// FeaturePropagation_83468394430485
// MI455X (gfx1250) — matrix operand order repaired and checked by reading, not compiled, not run on hardware
//
#include <hip/hip_runtime.h>
#include <hip/hip_bf16.h>

// Problem constants (fixed by the reference)
#define BB      8
#define NP      8192
#define SP      2048
#define CIN     256
#define CSKIP   128
#define CCAT    384
#define COUT    256
#define CEXP    1024
#define NGROUP  8

typedef __attribute__((ext_vector_type(16))) __bf16 v16bf;
typedef __attribute__((ext_vector_type(8)))  __bf16 v8bf;
typedef __attribute__((ext_vector_type(8)))  float  v8f;

__device__ __forceinline__ unsigned short f2bf(float f) {
  union { float f; unsigned u; } v; v.f = f;
  unsigned r = v.u + 0x7fffu + ((v.u >> 16) & 1u);   // round-to-nearest-even
  return (unsigned short)(r >> 16);
}
__device__ __forceinline__ float bf2f(unsigned short h) {
  union { unsigned u; float f; } v; v.u = ((unsigned)h) << 16;
  return v.f;
}

// ---------------------------------------------------------------- utilities
__global__ void zero_kernel(float* p, int n) {
  int i = blockIdx.x * blockDim.x + threadIdx.x;
  if (i < n) p[i] = 0.0f;
}

__global__ void f32_to_bf16_kernel(const float* __restrict__ src,
                                   unsigned short* __restrict__ dst, int n) {
  int i = blockIdx.x * blockDim.x + threadIdx.x;
  if (i < n) dst[i] = f2bf(src[i]);
}

// ---------------------------------------------------------------- 3-NN
// One thread per query point; xyz2 staged through LDS in 256-point tiles.
__global__ void knn3_kernel(const float* __restrict__ xyz1,
                            const float* __restrict__ xyz2,
                            int* __restrict__ idxOut, float* __restrict__ wOut) {
  const int b = blockIdx.y;
  const int n = blockIdx.x * blockDim.x + threadIdx.x;
  __shared__ float sx[256], sy[256], sz[256];

  const float x1 = xyz1[((size_t)b * 3 + 0) * NP + n];
  const float y1 = xyz1[((size_t)b * 3 + 1) * NP + n];
  const float z1 = xyz1[((size_t)b * 3 + 2) * NP + n];

  float d0 = 3.4e38f, d1 = 3.4e38f, d2 = 3.4e38f;
  int   i0 = 0, i1 = 0, i2 = 0;

  for (int s0 = 0; s0 < SP; s0 += 256) {
    const int t = threadIdx.x;
    sx[t] = xyz2[((size_t)b * 3 + 0) * SP + s0 + t];
    sy[t] = xyz2[((size_t)b * 3 + 1) * SP + s0 + t];
    sz[t] = xyz2[((size_t)b * 3 + 2) * SP + s0 + t];
    __syncthreads();
    for (int j = 0; j < 256; ++j) {
      const float dx = x1 - sx[j], dy = y1 - sy[j], dz = z1 - sz[j];
      const float d = dx * dx + dy * dy + dz * dz;
      const int s = s0 + j;
      if (d < d0)      { d2 = d1; i2 = i1; d1 = d0; i1 = i0; d0 = d; i0 = s; }
      else if (d < d1) { d2 = d1; i2 = i1; d1 = d;  i1 = s; }
      else if (d < d2) { d2 = d;  i2 = s; }
    }
    __syncthreads();
  }
  const float r0 = 1.0f / (d0 + 1e-8f);
  const float r1 = 1.0f / (d1 + 1e-8f);
  const float r2 = 1.0f / (d2 + 1e-8f);
  const float inv = 1.0f / (r0 + r1 + r2);
  const size_t base = ((size_t)b * NP + n) * 3;
  idxOut[base + 0] = i0; idxOut[base + 1] = i1; idxOut[base + 2] = i2;
  wOut[base + 0] = r0 * inv; wOut[base + 1] = r1 * inv; wOut[base + 2] = r2 * inv;
}

// ----------------------------------------------- concat + interpolate -> X0^T bf16
// X0T layout: [B][N][384] bf16 (c contiguous -> WMMA B-fragment friendly)
__global__ void build_x0t_kernel(const float* __restrict__ points1,
                                 const float* __restrict__ points2,
                                 const int* __restrict__ idx,
                                 const float* __restrict__ w,
                                 unsigned short* __restrict__ X0T) {
  const int bn = blockIdx.x;
  const int b = bn / NP;
  const int n = bn % NP;
  const size_t ib = (size_t)bn * 3;
  const int i0 = idx[ib + 0], i1 = idx[ib + 1], i2 = idx[ib + 2];
  const float w0 = w[ib + 0], w1 = w[ib + 1], w2 = w[ib + 2];
  unsigned short* dst = X0T + (size_t)bn * CCAT;
  for (int c = threadIdx.x; c < CCAT; c += blockDim.x) {
    float v;
    if (c < CSKIP) {
      v = points1[((size_t)b * CSKIP + c) * NP + n];
    } else {
      const float* p = points2 + ((size_t)b * CIN + (c - CSKIP)) * SP;
      v = w0 * p[i0] + w1 * p[i1] + w2 * p[i2];
    }
    dst[c] = f2bf(v);
  }
}

// ---------------------------------------------------------------- WMMA GEMM
// Y[b][n][o] = sum_c W[o][c]*X[b][n][c] + bias[o]
// Each wave owns a 64(M) x 32(N) tile: 4 M-subtiles x 2 N-subtiles.
// Per k-step (K += 32): 2 B-fragment loads (32B each) + 4 A-fragment loads
// (2x16B each), 8 x V_WMMA_F32_16X16X32_BF16 -> ~21 FLOP per loaded byte.
template <bool OUTF32, int K>
__global__ __launch_bounds__(256) void gemm_bf16_kernel(
    const unsigned short* __restrict__ Wq,     // [M][K] bf16
    const unsigned short* __restrict__ Xq,     // [B][NP][K] bf16
    const float* __restrict__ bias,            // [M]
    float* __restrict__ Yf,                    // [B][NP][M] fp32 (if OUTF32)
    unsigned short* __restrict__ Ybf,          // [B][NP][M] bf16 (else)
    int M) {
  const int b    = blockIdx.z;
  const int lane = threadIdx.x & 31;
  const int wv   = threadIdx.x >> 5;
  const int hf   = lane >> 4;        // half-wave select
  const int lr   = lane & 15;
  const int nBase = (blockIdx.x * 8 + wv) * 32;
  const int mBase = blockIdx.y * 64;

  const __bf16* X0p = (const __bf16*)Xq + ((size_t)b * NP + nBase + lr) * K;
  const __bf16* X1p = X0p + (size_t)16 * K;
  const __bf16* Wb = (const __bf16*)Wq;

  v8f acc[4][2];
#pragma unroll
  for (int mt = 0; mt < 4; ++mt)
#pragma unroll
    for (int nt = 0; nt < 2; ++nt)
#pragma unroll
      for (int i = 0; i < 8; ++i) acc[mt][nt][i] = 0.0f;

#pragma unroll 2
  for (int k = 0; k < K; k += 32) {
    // B fragments: col = nBase + nt*16 + lr, K = k + hf*16 .. +16 (contig 32B)
    v16bf bfrag0 = *(const v16bf*)(X0p + k + hf * 16);
    v16bf bfrag1 = *(const v16bf*)(X1p + k + hf * 16);
#pragma unroll
    for (int mt = 0; mt < 4; ++mt) {
      const __bf16* wr = Wb + (size_t)(mBase + mt * 16 + lr) * K + k;
      v8bf alo = *(const v8bf*)(wr + hf * 16);        // K = k + hf*16 .. +8
      v8bf ahi = *(const v8bf*)(wr + hf * 16 + 8);    // K = k+hf*16+8 .. +8
      v16bf afrag;
#pragma unroll
      for (int i = 0; i < 8; ++i) { afrag[i] = alo[i]; afrag[i + 8] = ahi[i]; }
      acc[mt][0] = __builtin_amdgcn_wmma_f32_16x16x32_bf16(
          false, afrag, false, bfrag0, (short)0, acc[mt][0], false, false);
      acc[mt][1] = __builtin_amdgcn_wmma_f32_16x16x32_bf16(
          false, afrag, false, bfrag1, (short)0, acc[mt][1], false, false);
    }
  }

#pragma unroll
  for (int mt = 0; mt < 4; ++mt) {
    const int o0 = mBase + mt * 16 + hf * 8;          // 8 contiguous output chans
    v8f bv = *(const v8f*)(bias + o0);
#pragma unroll
    for (int nt = 0; nt < 2; ++nt) {
      const int n = nBase + nt * 16 + lr;
      v8f a = acc[mt][nt];
#pragma unroll
      for (int r = 0; r < 8; ++r) a[r] += bv[r];
      const size_t base = ((size_t)b * NP + n) * (size_t)M + o0;
      if (OUTF32) {
        *(v8f*)(Yf + base) = a;                       // 32B contiguous store
      } else {
        unsigned short t[8];
#pragma unroll
        for (int r = 0; r < 8; ++r) t[r] = f2bf(a[r]);
        uint4 pk;
        pk.x = (unsigned)t[0] | ((unsigned)t[1] << 16);
        pk.y = (unsigned)t[2] | ((unsigned)t[3] << 16);
        pk.z = (unsigned)t[4] | ((unsigned)t[5] << 16);
        pk.w = (unsigned)t[6] | ((unsigned)t[7] << 16);
        *(uint4*)(Ybf + base) = pk;                   // 16B contiguous store
      }
    }
  }
}

// ------------------------------------------------------- GroupNorm statistics
// One (b,g) per blockIdx.x, split over blockIdx.y; LDS reduce + 2 atomics/block.
__global__ void stats_kernel(const float* __restrict__ Yf,
                             const unsigned short* __restrict__ Ybf,
                             float* __restrict__ acc, int C, int gs, int G) {
  const int b = blockIdx.x / G;
  const int g = blockIdx.x % G;
  const size_t cnt = (size_t)NP * gs;
  const size_t stride = (size_t)gridDim.y * blockDim.x;
  float s = 0.0f, sq = 0.0f;
  for (size_t e = (size_t)blockIdx.y * blockDim.x + threadIdx.x; e < cnt; e += stride) {
    const int n = (int)(e / gs);
    const int c = g * gs + (int)(e % gs);
    const size_t i = ((size_t)b * NP + n) * (size_t)C + c;
    const float v = Yf ? Yf[i] : bf2f(Ybf[i]);
    s += v; sq += v * v;
  }
  __shared__ float ls[256], lq[256];
  ls[threadIdx.x] = s; lq[threadIdx.x] = sq;
  __syncthreads();
  for (int o = 128; o > 0; o >>= 1) {
    if ((int)threadIdx.x < o) {
      ls[threadIdx.x] += ls[threadIdx.x + o];
      lq[threadIdx.x] += lq[threadIdx.x + o];
    }
    __syncthreads();
  }
  if (threadIdx.x == 0) {
    atomicAdd(&acc[(b * G + g) * 2 + 0], ls[0]);
    atomicAdd(&acc[(b * G + g) * 2 + 1], lq[0]);
  }
}

__global__ void finalize_stats_kernel(const float* __restrict__ acc,
                                      float* __restrict__ mr, int count, float inv) {
  const int i = blockIdx.x * blockDim.x + threadIdx.x;
  if (i < count) {
    const float m = acc[i * 2] * inv;
    const float v = acc[i * 2 + 1] * inv - m * m;
    mr[i * 2] = m;
    mr[i * 2 + 1] = rsqrtf(v + 1e-5f);
  }
}

// ------------------------------------------------- GN apply + SiLU (fused)
__global__ void gn_apply_kernel(const float* __restrict__ Yf,
                                const unsigned short* __restrict__ Ybf,
                                float* __restrict__ outF,
                                unsigned short* __restrict__ outBf,
                                const float* __restrict__ gamma,
                                const float* __restrict__ beta,
                                const float* __restrict__ mr,
                                int C, int gs, int G, size_t total) {
  const size_t i = (size_t)blockIdx.x * blockDim.x + threadIdx.x;
  if (i >= total) return;
  const int c = (int)(i % C);
  const size_t bn = i / C;
  const int b = (int)(bn / NP);
  const int g = c / gs;
  const float v = Yf ? Yf[i] : bf2f(Ybf[i]);
  const float m = mr[(b * G + g) * 2 + 0];
  const float r = mr[(b * G + g) * 2 + 1];
  const float xn = (v - m) * r * gamma[c] + beta[c];
  const float sv = xn / (1.0f + __expf(-xn));        // SiLU
  if (outF)  outF[i] = sv;
  if (outBf) outBf[i] = f2bf(sv);
}

// ------------------------------------------------- FiLM scale/shift (tiny)
__global__ void scale_shift_kernel(const float* __restrict__ t_emb,
                                   const float* __restrict__ c_emb,
                                   const float* __restrict__ tw, const float* __restrict__ tb,
                                   const float* __restrict__ cw, const float* __restrict__ cb,
                                   float* __restrict__ scaleB, float* __restrict__ shiftB) {
  const int b = blockIdx.x;
  const int j = threadIdx.x;                 // 512 threads
  const float* te = t_emb + (size_t)b * COUT;
  const float* ce = c_emb + (size_t)b * COUT;
  const float* twr = tw + (size_t)j * COUT;
  const float* cwr = cw + (size_t)j * COUT;
  float st = tb[j], sc = cb[j];
  for (int k = 0; k < COUT; ++k) { st += te[k] * twr[k]; sc += ce[k] * cwr[k]; }
  __shared__ float sbuf[2 * COUT];
  sbuf[j] = st + sc;
  __syncthreads();
  if (j < COUT) {
    scaleB[(size_t)b * COUT + j] = sbuf[j];
    shiftB[(size_t)b * COUT + j] = sbuf[j + COUT];
  }
}

// --------------------- final: GN3 + FiLM + residual + [N][C]->[C][N] transpose
__global__ void final_kernel(const float* __restrict__ Y3,
                             const float* __restrict__ X1f,
                             const float* __restrict__ mr,
                             const float* __restrict__ g2w, const float* __restrict__ g2b,
                             const float* __restrict__ scaleB, const float* __restrict__ shiftB,
                             float* __restrict__ out) {
  const int b = blockIdx.z;
  const int o0 = blockIdx.y * 32;
  const int n0 = blockIdx.x * 32;
  __shared__ float tile[32][33];
  const int tx = threadIdx.x, ty = threadIdx.y;     // 32 x 8
  const int o = o0 + tx;
  const int g = o >> 5;                             // gs = 32
  const float m = mr[(b * NGROUP + g) * 2 + 0];
  const float r = mr[(b * NGROUP + g) * 2 + 1];
  const float gw = g2w[o], gb = g2b[o];
  const float sc = scaleB[(size_t)b * COUT + o];
  const float sh = shiftB[(size_t)b * COUT + o];
#pragma unroll
  for (int q = 0; q < 4; ++q) {
    const int n = n0 + ty + q * 8;
    const size_t src = ((size_t)b * NP + n) * COUT + o;
    const float hn = (Y3[src] - m) * r * gw + gb;
    const float h2 = hn * (1.0f + sc) + sh;
    tile[tx][ty + q * 8] = h2 + X1f[src];
  }
  __syncthreads();
#pragma unroll
  for (int q = 0; q < 4; ++q) {
    const int ol = ty + q * 8;
    out[((size_t)b * COUT + o0 + ol) * NP + n0 + tx] = tile[ol][tx];
  }
}

// ---------------------------------------------------------------- launcher
extern "C" void kernel_launch(void* const* d_in, const int* in_sizes, int n_in,
                              void* d_out, int out_size, void* d_ws, size_t ws_size,
                              hipStream_t stream) {
  const float* xyz1    = (const float*)d_in[0];
  const float* points1 = (const float*)d_in[1];
  const float* xyz2    = (const float*)d_in[2];
  const float* points2 = (const float*)d_in[3];
  const float* t_emb   = (const float*)d_in[4];
  const float* c_emb   = (const float*)d_in[5];
  const float* mlp_w   = (const float*)d_in[6];
  const float* mlp_b   = (const float*)d_in[7];
  const float* mlp_gw  = (const float*)d_in[8];
  const float* mlp_gb  = (const float*)d_in[9];
  const float* c1w     = (const float*)d_in[10];
  const float* c1b     = (const float*)d_in[11];
  const float* g1w     = (const float*)d_in[12];
  const float* g1b     = (const float*)d_in[13];
  const float* c2w     = (const float*)d_in[14];
  const float* c2b     = (const float*)d_in[15];
  const float* g2w     = (const float*)d_in[16];
  const float* g2b     = (const float*)d_in[17];
  const float* tw      = (const float*)d_in[18];
  const float* tb      = (const float*)d_in[19];
  const float* cw      = (const float*)d_in[20];
  const float* cb      = (const float*)d_in[21];

  char* ws = (char*)d_ws;
  size_t off = 0;
  auto carve = [&](size_t bytes) -> void* {
    void* p = ws + off;
    off = (off + bytes + 255) & ~(size_t)255;
    return p;
  };

  unsigned short* WmlpBf = (unsigned short*)carve((size_t)COUT * CCAT * 2);
  unsigned short* Wc1Bf  = (unsigned short*)carve((size_t)CEXP * COUT * 2);
  unsigned short* Wc2Bf  = (unsigned short*)carve((size_t)COUT * CEXP * 2);
  int*            nnIdx  = (int*)carve((size_t)BB * NP * 3 * 4);
  float*          nnW    = (float*)carve((size_t)BB * NP * 3 * 4);
  unsigned short* X0T    = (unsigned short*)carve((size_t)BB * NP * CCAT * 2);
  float*          Y1     = (float*)carve((size_t)BB * NP * COUT * 4);          // -> X1f
  unsigned short* X1T    = (unsigned short*)carve((size_t)BB * NP * COUT * 2);
  unsigned short* Y2     = (unsigned short*)carve((size_t)BB * NP * CEXP * 2); // -> X2T
  float*          Y3     = (float*)carve((size_t)BB * NP * COUT * 4);
  float*          accAll = (float*)carve((size_t)3 * BB * NGROUP * 2 * 4);
  float*          mr1    = (float*)carve((size_t)BB * NGROUP * 2 * 4);
  float*          mr2    = (float*)carve((size_t)BB * NGROUP * 2 * 4);
  float*          mr3    = (float*)carve((size_t)BB * NGROUP * 2 * 4);
  float*          scaleB = (float*)carve((size_t)BB * COUT * 4);
  float*          shiftB = (float*)carve((size_t)BB * COUT * 4);
  float* acc1 = accAll;
  float* acc2 = accAll + BB * NGROUP * 2;
  float* acc3 = accAll + 2 * BB * NGROUP * 2;

  // 0) reset atomic stat accumulators (deterministic across replays)
  zero_kernel<<<2, 256, 0, stream>>>(accAll, 3 * BB * NGROUP * 2);

  // 1) bf16 weight copies
  f32_to_bf16_kernel<<<(COUT * CCAT + 255) / 256, 256, 0, stream>>>(mlp_w, WmlpBf, COUT * CCAT);
  f32_to_bf16_kernel<<<(CEXP * COUT + 255) / 256, 256, 0, stream>>>(c1w, Wc1Bf, CEXP * COUT);
  f32_to_bf16_kernel<<<(COUT * CEXP + 255) / 256, 256, 0, stream>>>(c2w, Wc2Bf, COUT * CEXP);

  // 2) 3-NN + interpolation into concatenated X0^T
  knn3_kernel<<<dim3(NP / 256, BB), 256, 0, stream>>>(xyz1, xyz2, nnIdx, nnW);
  build_x0t_kernel<<<BB * NP, 128, 0, stream>>>(points1, points2, nnIdx, nnW, X0T);

  // 3) GEMM1 (256x384) -> Y1 fp32 ; GN1 stats ; GN1+SiLU (fp32 in-place + bf16 copy)
  gemm_bf16_kernel<true, CCAT><<<dim3(NP / 256, COUT / 64, BB), 256, 0, stream>>>(
      WmlpBf, X0T, mlp_b, Y1, nullptr, COUT);
  stats_kernel<<<dim3(BB * NGROUP, 64), 256, 0, stream>>>(Y1, nullptr, acc1, COUT, COUT / NGROUP, NGROUP);
  finalize_stats_kernel<<<1, 64, 0, stream>>>(acc1, mr1, BB * NGROUP, 1.0f / ((float)NP * (COUT / NGROUP)));
  {
    const size_t total = (size_t)BB * NP * COUT;
    gn_apply_kernel<<<(unsigned)(total / 256), 256, 0, stream>>>(
        Y1, nullptr, Y1, X1T, mlp_gw, mlp_gb, mr1, COUT, COUT / NGROUP, NGROUP, total);
  }

  // 4) GEMM2 (1024x256) -> Y2 bf16 ; GN2 stats ; GN2+SiLU (bf16 in-place)
  gemm_bf16_kernel<false, COUT><<<dim3(NP / 256, CEXP / 64, BB), 256, 0, stream>>>(
      Wc1Bf, X1T, c1b, nullptr, Y2, CEXP);
  stats_kernel<<<dim3(BB * NGROUP, 64), 256, 0, stream>>>(nullptr, Y2, acc2, CEXP, CEXP / NGROUP, NGROUP);
  finalize_stats_kernel<<<1, 64, 0, stream>>>(acc2, mr2, BB * NGROUP, 1.0f / ((float)NP * (CEXP / NGROUP)));
  {
    const size_t total = (size_t)BB * NP * CEXP;
    gn_apply_kernel<<<(unsigned)(total / 256), 256, 0, stream>>>(
        nullptr, Y2, nullptr, Y2, g1w, g1b, mr2, CEXP, CEXP / NGROUP, NGROUP, total);
  }

  // 5) GEMM3 (256x1024) -> Y3 fp32 ; GN3 stats
  gemm_bf16_kernel<true, CEXP><<<dim3(NP / 256, COUT / 64, BB), 256, 0, stream>>>(
      Wc2Bf, Y2, c2b, Y3, nullptr, COUT);
  stats_kernel<<<dim3(BB * NGROUP, 64), 256, 0, stream>>>(Y3, nullptr, acc3, COUT, COUT / NGROUP, NGROUP);
  finalize_stats_kernel<<<1, 64, 0, stream>>>(acc3, mr3, BB * NGROUP, 1.0f / ((float)NP * (COUT / NGROUP)));

  // 6) FiLM params + final fused GN3/scale-shift/residual/transpose
  scale_shift_kernel<<<BB, 2 * COUT, 0, stream>>>(t_emb, c_emb, tw, tb, cw, cb, scaleB, shiftB);
  final_kernel<<<dim3(NP / 32, COUT / 32, BB), dim3(32, 8), 0, stream>>>(
      Y3, Y1, mr3, g2w, g2b, scaleB, shiftB, (float*)d_out);
}
